// TriangleProteinToCompound_14491219657353
// MI455X (gfx1250) — matrix operand order repaired and checked by reading, not compiled, not run on hardware
//
#include <hip/hip_runtime.h>
#include <hip/hip_bf16.h>
#include <stdint.h>

// ---------------------------------------------------------------------------
// TriangleProteinToCompound on MI455X (gfx1250, wave32, WMMA bf16 path)
//   E=256, C=128, NP=512, NC=128, rows R = NP*NC = 65536
// All GEMMs run on v_wmma_f32_16x16x32_bf16.  B operands are pre-packed into
// fragment order (32 lanes x 16 bf16 per 32x16 tile) so every fragment load
// is two 16-byte loads.  The triangle kernel register-blocks 32x32 outputs
// per wave (4 accumulators) for 4 WMMAs per 2 A-frag + 2 B-frag loads.
// ---------------------------------------------------------------------------

typedef __bf16 bf16_t;
typedef __bf16 v16bf __attribute__((ext_vector_type(16)));
typedef float  v8f   __attribute__((ext_vector_type(8)));

#define E_DIM 256
#define C_DIM 128
#define NP    512
#define NC    128
#define R_TOT (NP * NC)   // 65536
#define TI    64          // i-rows per triangle block

__device__ __forceinline__ bf16_t f2bf(float x) { return (bf16_t)x; }

// A fragment: 16x32 bf16 from row-major [M x K] panel, leading dim ld.
// ISA layout: lanes 0-15 hold K={0..7,16..23}; lanes 16-31 hold K={8..15,24..31}.
__device__ __forceinline__ v16bf load_frag_a(const bf16_t* __restrict__ p, int ld, int lane) {
  const int row = lane & 15;
  const int kh  = (lane >> 4) << 3;           // 0 or 8
  const bf16_t* r = p + row * ld + kh;
  v16bf a;
  *((uint4*)&a)     = *(const uint4*)(r);      // K = kh .. kh+7
  *(((uint4*)&a)+1) = *(const uint4*)(r + 16); // K = kh+16 .. kh+23
  return a;
}

// B fragment from fragment-packed panel: frag points at a 512-element tile,
// lane's 16 bf16 are contiguous at frag + lane*16.
__device__ __forceinline__ v16bf load_frag_b_packed(const bf16_t* __restrict__ frag, int lane) {
  const bf16_t* p = frag + lane * 16;
  v16bf b;
  *((uint4*)&b)     = *(const uint4*)(p);
  *(((uint4*)&b)+1) = *(const uint4*)(p + 8);
  return b;
}

__device__ __forceinline__ v8f wmma_bf16(v16bf a, v16bf b, v8f c) {
  return __builtin_amdgcn_wmma_f32_16x16x32_bf16(false, a, false, b, (short)0, c, false, false);
}

// Packed-fragment index: panel is [K x N]; fragment id = (k/32)*(N/16)+(n/16),
// within fragment: lane = ((k>>3)&1)<<4 | (n&15), elem = (k&7) | ((k&31)>>4)<<3.
__device__ __forceinline__ size_t pack_off(int k, int n, int N) {
  const int kk   = k & 31;
  const int lane = (((kk >> 3) & 1) << 4) | (n & 15);
  const int elem = (kk & 7) | ((kk >> 4) << 3);
  return ((size_t)(k >> 5) * (N >> 4) + (n >> 4)) * 512 + lane * 16 + elem;
}

// ---------------------------------------------------------------------------
// Kernel 0: fp32 [K x N] row-major weights -> fragment-packed bf16.
// ---------------------------------------------------------------------------
__global__ void k_pack_w(const float* __restrict__ src, bf16_t* __restrict__ dst,
                         int K, int N) {
  const int idx = blockIdx.x * 256 + threadIdx.x;
  if (idx >= K * N) return;
  const int k = idx / N, n = idx - k * N;
  dst[pack_off(k, n, N)] = f2bf(src[idx]);
}

// Kernel 0b: planar bf16 [plane][K x N] -> fragment-packed bf16 (ab panel).
__global__ void k_pack_plane(const bf16_t* __restrict__ src, bf16_t* __restrict__ dst,
                             int K, int N) {
  const size_t pbase = (size_t)blockIdx.y * K * N;
  const int idx = blockIdx.x * 256 + threadIdx.x;
  if (idx >= K * N) return;
  const int k = idx / N, n = idx - k * N;
  dst[pbase + pack_off(k, n, N)] = src[pbase + idx];
}

// ---------------------------------------------------------------------------
// Kernel 1: LayerNorm over E=256, one row per wave32, output bf16.
// ---------------------------------------------------------------------------
__global__ void __launch_bounds__(256) k_layernorm(const float* __restrict__ z,
                                                   const float* __restrict__ w,
                                                   const float* __restrict__ b,
                                                   bf16_t* __restrict__ out) {
  const int lane = threadIdx.x & 31;
  const int wave = threadIdx.x >> 5;
  const int row  = blockIdx.x * 8 + wave;
  const float* zr = z + (size_t)row * E_DIM;
  float x[8], s = 0.f, sq = 0.f;
#pragma unroll
  for (int t = 0; t < 8; ++t) {
    x[t] = zr[lane + 32 * t];
    s += x[t]; sq += x[t] * x[t];
  }
#pragma unroll
  for (int off = 16; off > 0; off >>= 1) {
    s  += __shfl_xor(s,  off, 32);
    sq += __shfl_xor(sq, off, 32);
  }
  const float mean = s * (1.f / E_DIM);
  const float var  = sq * (1.f / E_DIM) - mean * mean;
  const float rinv = rsqrtf(var + 1e-5f);
  bf16_t* orow = out + (size_t)row * E_DIM;
#pragma unroll
  for (int t = 0; t < 8; ++t) {
    const int e = lane + 32 * t;
    orow[e] = f2bf((x[t] - mean) * rinv * w[e] + b[e]);
  }
}

// ---------------------------------------------------------------------------
// Kernel 2: ab = sigmoid(zn@gate_w + gb) * (zn@lin_w + lb) * mask
//           -> planar bf16 abT[c][row].  Two fused WMMA GEMMs, K=256.
// ---------------------------------------------------------------------------
__global__ void __launch_bounds__(256) k_gate_lin(const bf16_t* __restrict__ zn,
                                                  const bf16_t* __restrict__ gwP,
                                                  const float* __restrict__ gb,
                                                  const bf16_t* __restrict__ lwP,
                                                  const float* __restrict__ lb,
                                                  const float* __restrict__ mask,
                                                  bf16_t* __restrict__ abT) {
  __shared__ __align__(16) bf16_t lA[16 * E_DIM];   // 8 KB A strip
  const int t = threadIdx.x, lane = t & 31, wave = t >> 5;
  const int row0 = blockIdx.x * 16;
  {
    const uint4* s4 = (const uint4*)(zn + (size_t)row0 * E_DIM);
    uint4* d4 = (uint4*)lA;
    for (int i = t; i < (16 * E_DIM) / 8; i += 256) d4[i] = s4[i];
  }
  __syncthreads();
  const int j0 = wave * 16;
  const int jt = j0 >> 4;
  v8f ag = {}, al = {};
#pragma unroll
  for (int kb = 0; kb < E_DIM; kb += 32) {
    const size_t fo = ((size_t)(kb >> 5) * (C_DIM >> 4) + jt) * 512;
    v16bf a  = load_frag_a(lA + kb, E_DIM, lane);
    v16bf bg = load_frag_b_packed(gwP + fo, lane);
    v16bf bl = load_frag_b_packed(lwP + fo, lane);
    ag = wmma_bf16(a, bg, ag);
    al = wmma_bf16(a, bl, al);
  }
  const int n = lane & 15, mb = (lane >> 4) << 3;
  const int c = j0 + n;
#pragma unroll
  for (int v = 0; v < 8; ++v) {
    const int row = row0 + mb + v;
    const float m   = mask[row];
    const float sig = 1.f / (1.f + __expf(-(ag[v] + gb[c])));
    abT[(size_t)c * R_TOT + row] = f2bf(sig * (al[v] + lb[c]) * m);
  }
}

// ---------------------------------------------------------------------------
// Kernel 3: g = sigmoid(zn @ eg_w + eg_b)  -> bf16 [row][e], K=256, N=256.
// ---------------------------------------------------------------------------
__global__ void __launch_bounds__(256) k_edge_gate(const bf16_t* __restrict__ zn,
                                                   const bf16_t* __restrict__ ewP,
                                                   const float* __restrict__ eb,
                                                   bf16_t* __restrict__ g) {
  __shared__ __align__(16) bf16_t lA[16 * E_DIM];
  const int t = threadIdx.x, lane = t & 31, wave = t >> 5;
  const int row0 = blockIdx.x * 16;
  {
    const uint4* s4 = (const uint4*)(zn + (size_t)row0 * E_DIM);
    uint4* d4 = (uint4*)lA;
    for (int i = t; i < (16 * E_DIM) / 8; i += 256) d4[i] = s4[i];
  }
  __syncthreads();
#pragma unroll
  for (int half = 0; half < 2; ++half) {
    const int e0 = (wave + 8 * half) * 16;
    v8f acc = {};
#pragma unroll
    for (int kb = 0; kb < E_DIM; kb += 32) {
      v16bf a = load_frag_a(lA + kb, E_DIM, lane);
      v16bf b = load_frag_b_packed(ewP + ((size_t)(kb >> 5) * (E_DIM >> 4) + (e0 >> 4)) * 512, lane);
      acc = wmma_bf16(a, b, acc);
    }
    const int n = lane & 15, mb = (lane >> 4) << 3;
    const int e = e0 + n;
#pragma unroll
    for (int v = 0; v < 8; ++v) {
      const int row = row0 + mb + v;
      g[(size_t)row * E_DIM + e] = f2bf(1.f / (1.f + __expf(-(acc[v] + eb[e]))));
    }
  }
}

// ---------------------------------------------------------------------------
// Kernel 4a: planarize [row][128c] fp32 -> [c][row] bf16 via LDS tile transpose.
// ---------------------------------------------------------------------------
__global__ void __launch_bounds__(256) k_planarize(const float* __restrict__ src,
                                                   bf16_t* __restrict__ dst,
                                                   int rows_total) {
  __shared__ float tile[32][129];
  const int t = threadIdx.x;
  const int row0 = blockIdx.x * 32;
  for (int idx = t; idx < 32 * 128; idx += 256) {
    const int r = idx >> 7, c = idx & 127;
    tile[r][c] = src[(size_t)(row0 + r) * 128 + c];
  }
  __syncthreads();
  for (int idx = t; idx < 32 * 128; idx += 256) {
    const int c = idx >> 5, r = idx & 31;
    dst[(size_t)c * rows_total + row0 + r] = f2bf(tile[r][c]);
  }
}

// Kernel 4b: compound_pair[j,k,c] -> packed B panels cpP[c]{k x j} (K=128,N=128).
__global__ void k_cp_pack(const float* __restrict__ cp, bf16_t* __restrict__ cpP) {
  const int c = threadIdx.x;   // 0..127
  const int k = blockIdx.x;    // 0..127
  const int j = blockIdx.y;    // 0..127
  const float v = cp[((size_t)j * 128 + k) * 128 + c];
  cpP[(size_t)c * (NC * NC) + pack_off(k, j, NC)] = f2bf(v);
}

// ---------------------------------------------------------------------------
// Kernel 5: per-channel triangle GEMMs, register-blocked 32x32 per wave.
//   blk[i,j,c] = sum_k ppT[c][i][k]*ab[c][k][j]  (K=512, B from packed abP)
//              + sum_k ab[c][i][k]*cp[c][k][j]   (K=128, B from packed cpP)
// Grid (i-block = NP/64, c=128); 8 waves as 2(i) x 4(j); each wave owns
// two 16-row i-tiles and two 16-col j-tiles -> 4 accumulators.
// ---------------------------------------------------------------------------
__global__ void __launch_bounds__(256) k_triangle(const bf16_t* __restrict__ ppT,
                                                  const bf16_t* __restrict__ abT,
                                                  const bf16_t* __restrict__ abP,
                                                  const bf16_t* __restrict__ cpP,
                                                  float* __restrict__ blk) {
  __shared__ __align__(16) bf16_t lA[TI * NP];   // 64 KB A strip (64 rows x 512)
  const int t = threadIdx.x, lane = t & 31, wave = t >> 5;
  const int i0 = blockIdx.x * TI;
  const int c  = blockIdx.y;
  const int iw = (wave >> 2) * 32;     // 0 or 32: this wave's 32-row slab
  const int j0 = (wave & 3) * 32;      // 0,32,64,96: this wave's 32-col slab
  const int jt = j0 >> 4;              // first j-tile index

  { // stage A strip (64 contiguous rows of 512)
    const uint4* s4 = (const uint4*)(ppT + (size_t)c * NP * NP + (size_t)i0 * NP);
    uint4* d4 = (uint4*)lA;
    for (int i = t; i < (TI * NP) / 8; i += 256) d4[i] = s4[i];
  }
  __syncthreads();

  const bf16_t* BP = abP + (size_t)c * R_TOT;    // packed [512 x 128] panel
  v8f acc00 = {}, acc01 = {}, acc10 = {}, acc11 = {};
#pragma unroll 4
  for (int kb = 0; kb < NP; kb += 32) {
    const bf16_t* fb = BP + ((size_t)(kb >> 5) * (NC >> 4) + jt) * 512;
    if (kb + 32 < NP)
      __builtin_prefetch(fb + (NC >> 4) * 512 + lane * 16, 0, 1);
    v16bf a0 = load_frag_a(lA + (iw     ) * NP + kb, NP, lane);
    v16bf a1 = load_frag_a(lA + (iw + 16) * NP + kb, NP, lane);
    v16bf b0 = load_frag_b_packed(fb,       lane);
    v16bf b1 = load_frag_b_packed(fb + 512, lane);
    acc00 = wmma_bf16(a0, b0, acc00);
    acc01 = wmma_bf16(a0, b1, acc01);
    acc10 = wmma_bf16(a1, b0, acc10);
    acc11 = wmma_bf16(a1, b1, acc11);
  }

  // block2: A = ab rows i0..i0+63 (planar, 64x128), B = packed cpP
  __syncthreads();
  {
    const uint4* s4 = (const uint4*)(abT + (size_t)c * R_TOT + (size_t)i0 * NC);
    uint4* d4 = (uint4*)lA;
    for (int i = t; i < (TI * NC) / 8; i += 256) d4[i] = s4[i];
  }
  __syncthreads();
  const bf16_t* CP = cpP + (size_t)c * NC * NC;
#pragma unroll
  for (int kb = 0; kb < NC; kb += 32) {
    const bf16_t* fb = CP + ((size_t)(kb >> 5) * (NC >> 4) + jt) * 512;
    v16bf a0 = load_frag_a(lA + (iw     ) * NC + kb, NC, lane);
    v16bf a1 = load_frag_a(lA + (iw + 16) * NC + kb, NC, lane);
    v16bf b0 = load_frag_b_packed(fb,       lane);
    v16bf b1 = load_frag_b_packed(fb + 512, lane);
    acc00 = wmma_bf16(a0, b0, acc00);
    acc01 = wmma_bf16(a0, b1, acc01);
    acc10 = wmma_bf16(a1, b0, acc10);
    acc11 = wmma_bf16(a1, b1, acc11);
  }

  // store 4 tiles (C/D layout: lane n = col, VGPR v -> row mb+v)
  const int n = lane & 15, mb = (lane >> 4) << 3;
  const int ja = j0 + n, jb = j0 + 16 + n;
#pragma unroll
  for (int v = 0; v < 8; ++v) {
    const int ia = i0 + iw + mb + v;
    const int ib = ia + 16;
    blk[((size_t)ia * NC + ja) * C_DIM + c] = acc00[v];
    blk[((size_t)ia * NC + jb) * C_DIM + c] = acc01[v];
    blk[((size_t)ib * NC + ja) * C_DIM + c] = acc10[v];
    blk[((size_t)ib * NC + jb) * C_DIM + c] = acc11[v];
  }
}

// ---------------------------------------------------------------------------
// Kernel 6: out = g * (LN_c(blk) @ as_w + as_b) * mask   (fp32 output)
// ---------------------------------------------------------------------------
__global__ void __launch_bounds__(256) k_out(const float* __restrict__ blk,
                                             const float* __restrict__ lnw,
                                             const float* __restrict__ lnb,
                                             const bf16_t* __restrict__ awP,
                                             const float* __restrict__ abias,
                                             const bf16_t* __restrict__ g,
                                             const float* __restrict__ mask,
                                             float* __restrict__ out) {
  __shared__ float  tileF[16 * C_DIM];                 // 8 KB
  __shared__ __align__(16) bf16_t lnA[16 * C_DIM];     // 4 KB
  __shared__ float mrow[16], rrow[16];
  const int t = threadIdx.x, lane = t & 31, wave = t >> 5;
  const int row0 = blockIdx.x * 16;

  for (int idx = t; idx < 16 * C_DIM; idx += 256)
    tileF[idx] = blk[(size_t)row0 * C_DIM + idx];
  __syncthreads();
  if (t < 16) {
    float s = 0.f, sq = 0.f;
    for (int c = 0; c < C_DIM; ++c) {
      const float x = tileF[t * C_DIM + c];
      s += x; sq += x * x;
    }
    const float mean = s * (1.f / C_DIM);
    const float var  = sq * (1.f / C_DIM) - mean * mean;
    mrow[t] = mean;
    rrow[t] = rsqrtf(var + 1e-5f);
  }
  __syncthreads();
  for (int idx = t; idx < 16 * C_DIM; idx += 256) {
    const int r = idx >> 7, c = idx & 127;
    lnA[idx] = f2bf((tileF[idx] - mrow[r]) * rrow[r] * lnw[c] + lnb[c]);
  }
  __syncthreads();

#pragma unroll
  for (int half = 0; half < 2; ++half) {
    const int e0 = (wave + 8 * half) * 16;
    v8f acc = {};
#pragma unroll
    for (int kb = 0; kb < C_DIM; kb += 32) {
      v16bf a = load_frag_a(lnA + kb, C_DIM, lane);
      v16bf b = load_frag_b_packed(awP + ((size_t)(kb >> 5) * (E_DIM >> 4) + (e0 >> 4)) * 512, lane);
      acc = wmma_bf16(a, b, acc);
    }
    const int n = lane & 15, mb = (lane >> 4) << 3;
    const int e = e0 + n;
#pragma unroll
    for (int v = 0; v < 8; ++v) {
      const int row = row0 + mb + v;
      const float gv = (float)g[(size_t)row * E_DIM + e];
      out[(size_t)row * E_DIM + e] = gv * (acc[v] + abias[e]) * mask[row];
    }
  }
}

// ---------------------------------------------------------------------------
extern "C" void kernel_launch(void* const* d_in, const int* in_sizes, int n_in,
                              void* d_out, int out_size, void* d_ws, size_t ws_size,
                              hipStream_t stream) {
  const float* z      = (const float*)d_in[0];
  const float* pp     = (const float*)d_in[1];
  const float* cp     = (const float*)d_in[2];
  const float* mask   = (const float*)d_in[3];
  const float* ln_w   = (const float*)d_in[4];
  const float* ln_b   = (const float*)d_in[5];
  const float* lnc_w  = (const float*)d_in[6];
  const float* lnc_b  = (const float*)d_in[7];
  const float* gate_w = (const float*)d_in[8];
  const float* gate_b = (const float*)d_in[9];
  const float* lin_w  = (const float*)d_in[10];
  const float* lin_b  = (const float*)d_in[11];
  const float* eg_w   = (const float*)d_in[12];
  const float* eg_b   = (const float*)d_in[13];
  const float* as_w   = (const float*)d_in[14];
  const float* as_b   = (const float*)d_in[15];
  float* out = (float*)d_out;

  char* ws = (char*)d_ws;
  size_t off = 0;
  auto carve = [&](size_t bytes) -> char* {
    char* p = ws + off;
    off = (off + bytes + 255) & ~(size_t)255;
    return p;
  };
  bf16_t* znb = (bf16_t*)carve((size_t)R_TOT * E_DIM * 2);   //  32 MB
  bf16_t* abT = (bf16_t*)carve((size_t)C_DIM * R_TOT * 2);   //  16 MB planar
  bf16_t* abP = (bf16_t*)carve((size_t)C_DIM * R_TOT * 2);   //  16 MB packed
  bf16_t* gbu = (bf16_t*)carve((size_t)R_TOT * E_DIM * 2);   //  32 MB
  bf16_t* ppT = (bf16_t*)carve((size_t)C_DIM * NP * NP * 2); //  64 MB
  bf16_t* cpP = (bf16_t*)carve((size_t)C_DIM * NC * NC * 2); //   4 MB packed
  float*  blk = (float*) carve((size_t)R_TOT * C_DIM * 4);   //  32 MB
  bf16_t* gwP = (bf16_t*)carve((size_t)E_DIM * C_DIM * 2);
  bf16_t* lwP = (bf16_t*)carve((size_t)E_DIM * C_DIM * 2);
  bf16_t* ewP = (bf16_t*)carve((size_t)E_DIM * E_DIM * 2);
  bf16_t* awP = (bf16_t*)carve((size_t)C_DIM * E_DIM * 2);

  // 0) weight packing (fragment order)
  k_pack_w<<<(E_DIM * C_DIM + 255) / 256, 256, 0, stream>>>(gate_w, gwP, E_DIM, C_DIM);
  k_pack_w<<<(E_DIM * C_DIM + 255) / 256, 256, 0, stream>>>(lin_w,  lwP, E_DIM, C_DIM);
  k_pack_w<<<(E_DIM * E_DIM + 255) / 256, 256, 0, stream>>>(eg_w,   ewP, E_DIM, E_DIM);
  k_pack_w<<<(C_DIM * E_DIM + 255) / 256, 256, 0, stream>>>(as_w,   awP, C_DIM, E_DIM);

  // 1) layernorm over E
  k_layernorm<<<R_TOT / 8, 256, 0, stream>>>(z, ln_w, ln_b, znb);

  // 2) ab (planar) then pack into B-fragment order; 3) edge gate g
  k_gate_lin<<<R_TOT / 16, 256, 0, stream>>>(znb, gwP, gate_b, lwP, lin_b, mask, abT);
  k_pack_plane<<<dim3((NP * NC) / 256, C_DIM), 256, 0, stream>>>(abT, abP, NP, NC);
  k_edge_gate<<<R_TOT / 16, 256, 0, stream>>>(znb, ewP, eg_b, gbu);

  // 4) planarize protein pair; pack compound pair
  k_planarize<<<(NP * NP) / 32, 256, 0, stream>>>(pp, ppT, NP * NP);
  k_cp_pack<<<dim3(NC, NC), 128, 0, stream>>>(cp, cpP);

  // 5) per-channel triangle WMMA GEMMs (register-blocked)
  k_triangle<<<dim3(NP / TI, C_DIM), 256, 0, stream>>>(ppT, abT, abP, cpP, blk);

  // 6) LN over C + final GEMM + gating
  k_out<<<R_TOT / 16, 256, 0, stream>>>(blk, lnc_w, lnc_b, awP, as_b, gbu, mask, out);

  (void)in_sizes; (void)n_in; (void)out_size; (void)ws_size;
}
